// LocalMapAwareRouteGenerator_18794776888007
// MI455X (gfx1250) — matrix operand order repaired and checked by reading, not compiled, not run on hardware
//
#include <hip/hip_runtime.h>
#include <hip/hip_bf16.h>

// ---------------------------------------------------------------------------
// LocalMapAwareRouteGenerator for MI455X (gfx1250, wave32, WMMA).
// All matrix math uses V_WMMA_F32_16X16X32_F16 (f16 inputs, f32 accumulate).
// Register-blocked: each wave computes 1 M-tile x {1..8} N-tiles so one A
// fragment feeds multiple WMMAs (HBM-bound workload -> maximize reuse).
// BN batch-stats via deterministic two-stage LDS reductions; LSTM runs in one
// workgroup with LDS-resident state and WMMA gate GEMMs.
// ---------------------------------------------------------------------------

typedef __attribute__((ext_vector_type(16))) _Float16 v16h;
typedef __attribute__((ext_vector_type(8)))  _Float16 v8h;
typedef __attribute__((ext_vector_type(8)))  float    v8f;

#define NCROP 512      // B*S
#define SEQ   64
#define BATCH 8

// ---- WMMA fragment helpers (ISA 7.12.2 layouts, wave32) -------------------
// A (16x32 f16): lane l holds row M=l%16; element e (vgpr v=e/2, half h=e%2)
//   K(l,e) = 16*(e>=8) + 8*(l/16) + (e&7)
// B (32x16 f16), stored as weights[N][K] row-major: lane l holds col N=l%16;
//   K(l,e) = 16*(e>=8) + 8*(l/16) + (e&7)   (same as A)
// C/D (16x16 f32): element r: M = 8*(l/16)+r, N = l%16.

__device__ __forceinline__ v16h load_a16(const _Float16* A, int lda, int m0,
                                         int k0, int lane) {
  int m = m0 + (lane & 15);
  int g = lane >> 4;
  const _Float16* p = A + (size_t)m * lda + k0 + g * 8;
  v8h lo = *(const v8h*)(p);        // K = k0 + g*8 + [0..7]
  v8h hi = *(const v8h*)(p + 16);   // K = k0 + 16 + g*8 + [0..7]
  v16h r;
#pragma unroll
  for (int e = 0; e < 8; ++e) { r[e] = lo[e]; r[8 + e] = hi[e]; }
  return r;
}

__device__ __forceinline__ v16h load_b16(const _Float16* W, int ldw, int n0,
                                         int k0, int lane) {
  int n = n0 + (lane & 15);
  int g = lane >> 4;
  const _Float16* p = W + (size_t)n * ldw + k0 + g * 8;
  v8h lo = *(const v8h*)(p);
  v8h hi = *(const v8h*)(p + 16);
  v16h r;
#pragma unroll
  for (int e = 0; e < 8; ++e) { r[e] = lo[e]; r[8 + e] = hi[e]; }
  return r;
}

__device__ __forceinline__ v8f wmma_16x16x32(v16h a, v16h b, v8f c) {
  return __builtin_amdgcn_wmma_f32_16x16x32_f16(false, a, false, b,
                                                (short)0, c, false, false);
}

__device__ __forceinline__ float sigmoidf(float x) {
  return 1.0f / (1.0f + __expf(-x));
}

// ---------------------------------------------------------------------------
// Crop gather: padded-by-64 dynamic-slice crops -> NHWC f16 [512][128][128][3]
// ---------------------------------------------------------------------------
__global__ __launch_bounds__(256) void crop_kernel(
    const float* __restrict__ full_map, const float* __restrict__ input_seq,
    _Float16* __restrict__ crops) {
  size_t idx = (size_t)blockIdx.x * 256 + threadIdx.x;
  const size_t total = (size_t)NCROP * 128 * 128 * 3;
  if (idx >= total) return;
  int ci = (int)(idx % 3);
  size_t t = idx / 3;
  int x = (int)(t % 128); t /= 128;
  int y = (int)(t % 128);
  int bs = (int)(t / 128);
  int b = bs >> 6;
  int y0 = (int)input_seq[(size_t)bs * 2 + 0];
  int x0 = (int)input_seq[(size_t)bs * 2 + 1];
  int sy = y0 + y - 64, sx = x0 + x - 64;
  float v = 0.f;
  if ((unsigned)sy < 1024u && (unsigned)sx < 1024u)
    v = full_map[(((size_t)b * 3 + ci) * 1024 + sy) * 1024 + sx];
  crops[idx] = (_Float16)v;
}

// ---------------------------------------------------------------------------
// Weight prep: conv OIHW f32 -> [O][KPAD] f16 with K = (ky*3+kx)*I + ci
// ---------------------------------------------------------------------------
template <int I, int O>
__global__ __launch_bounds__(256) void prep_conv_w(const float* __restrict__ cw,
                                                   _Float16* __restrict__ wp) {
  constexpr int K = 9 * I;
  constexpr int KPAD = (K + 31) & ~31;
  int idx = blockIdx.x * 256 + threadIdx.x;
  if (idx >= O * KPAD) return;
  int o = idx / KPAD, k = idx % KPAD;
  _Float16 v = (_Float16)0.f;
  if (k < K) {
    int kyx = k / I, ci = k - kyx * I;
    int ky = kyx / 3, kx = kyx - ky * 3;
    v = (_Float16)cw[((o * I + ci) * 3 + ky) * 3 + kx];
  }
  wp[idx] = v;
}

// fw1 (512,8192 c-major NCHW flatten) -> [512][8192] with NHWC k = (y*8+x)*128+c
__global__ __launch_bounds__(256) void prep_fc1(const float* __restrict__ fw1,
                                                _Float16* __restrict__ wp) {
  int idx = blockIdx.x * 256 + threadIdx.x;
  if (idx >= 512 * 8192) return;
  int o = idx >> 13, kn = idx & 8191;
  int c = kn & 127, yx = kn >> 7;
  wp[idx] = (_Float16)fw1[((size_t)o << 13) + (c << 6) + yx];
}

__global__ __launch_bounds__(256) void cast_f16_kernel(
    const float* __restrict__ src, _Float16* __restrict__ dst, size_t n) {
  size_t i = (size_t)blockIdx.x * 256 + threadIdx.x;
  if (i < n) dst[i] = (_Float16)src[i];
}

// ---------------------------------------------------------------------------
// Implicit-GEMM conv (3x3, stride 2, pad 1), NHWC f16 in, f32 raw out (pre-BN).
// One wave: 16 output pixels x ALL O channels (NT accumulators) so the
// bounds-checked im2col gather is done once and feeds NT WMMAs per k-step.
// ---------------------------------------------------------------------------
template <int HIN, int I, int O>
__global__ __launch_bounds__(256) void conv_wmma(
    const _Float16* __restrict__ in, const _Float16* __restrict__ wp,
    const float* __restrict__ cb, float* __restrict__ outraw) {
  constexpr int WIN = HIN;
  constexpr int HOUT = HIN / 2, WOUT = HIN / 2;
  constexpr int K = 9 * I;
  constexpr int KPAD = (K + 31) & ~31;
  constexpr int NT = O / 16;
  constexpr int MPIX = HOUT * WOUT;
  constexpr int MTOT = NCROP * MPIX;
  int wave = (blockIdx.x * 256 + threadIdx.x) >> 5;
  int lane = threadIdx.x & 31;
  int mtile = wave;
  if (mtile * 16 >= MTOT) return;
  int g = lane >> 4;
  int m = mtile * 16 + (lane & 15);
  int crop = m / MPIX;
  int p = m - crop * MPIX;
  int oy = p / WOUT, ox = p - oy * WOUT;
  const _Float16* base = in + (size_t)crop * HIN * WIN * I;
  v8f acc[NT] = {};
  for (int k0 = 0; k0 < KPAD; k0 += 32) {
    v16h a;
#pragma unroll
    for (int e = 0; e < 16; ++e) {
      int k = k0 + ((e >> 3) << 4) + g * 8 + (e & 7);
      _Float16 val = (_Float16)0.f;
      if (k < K) {
        int kyx = k / I, ci = k - kyx * I;
        int ky = kyx / 3, kx = kyx - ky * 3;
        int iy = oy * 2 - 1 + ky, ix = ox * 2 - 1 + kx;
        if ((unsigned)iy < (unsigned)HIN && (unsigned)ix < (unsigned)WIN)
          val = base[((size_t)iy * WIN + ix) * I + ci];
      }
      a[e] = val;
    }
#pragma unroll
    for (int nt = 0; nt < NT; ++nt) {
      v16h b = load_b16(wp, KPAD, nt * 16, k0, lane);
      acc[nt] = wmma_16x16x32(a, b, acc[nt]);
    }
  }
#pragma unroll
  for (int nt = 0; nt < NT; ++nt) {
    int n = nt * 16 + (lane & 15);
    float bias = cb[n];
#pragma unroll
    for (int r = 0; r < 8; ++r) {
      int m2 = mtile * 16 + g * 8 + r;
      outraw[(size_t)m2 * O + n] = acc[nt][r] + bias;  // pre-BN value
    }
  }
}

// ---------------------------------------------------------------------------
// BatchNorm (training-mode batch stats) — deterministic two-stage reduction
// ---------------------------------------------------------------------------
__global__ __launch_bounds__(256) void bn_stats_partial(
    const float* __restrict__ y, int Mtot, int O, float* __restrict__ partial) {
  __shared__ float s1[256], s2[256];
  int c = blockIdx.y;
  int chunk = blockIdx.x;
  int nchunks = gridDim.x;
  int per = (Mtot + nchunks - 1) / nchunks;
  int start = chunk * per;
  int end = start + per; if (end > Mtot) end = Mtot;
  float a = 0.f, b = 0.f;
  for (int i = start + (int)threadIdx.x; i < end; i += 256) {
    float v = y[(size_t)i * O + c];
    a += v; b += v * v;
  }
  s1[threadIdx.x] = a; s2[threadIdx.x] = b;
  __syncthreads();
  for (int st = 128; st > 0; st >>= 1) {
    if ((int)threadIdx.x < st) {
      s1[threadIdx.x] += s1[threadIdx.x + st];
      s2[threadIdx.x] += s2[threadIdx.x + st];
    }
    __syncthreads();
  }
  if (threadIdx.x == 0) {
    partial[(size_t)(chunk * O + c) * 2 + 0] = s1[0];
    partial[(size_t)(chunk * O + c) * 2 + 1] = s2[0];
  }
}

__global__ __launch_bounds__(256) void bn_stats_final(
    const float* __restrict__ partial, int O, int Mtot,
    float* __restrict__ stats) {
  __shared__ float s1[256], s2[256];
  int c = blockIdx.x;
  s1[threadIdx.x] = partial[(size_t)(threadIdx.x * O + c) * 2 + 0];
  s2[threadIdx.x] = partial[(size_t)(threadIdx.x * O + c) * 2 + 1];
  __syncthreads();
  for (int st = 128; st > 0; st >>= 1) {
    if ((int)threadIdx.x < st) {
      s1[threadIdx.x] += s1[threadIdx.x + st];
      s2[threadIdx.x] += s2[threadIdx.x + st];
    }
    __syncthreads();
  }
  if (threadIdx.x == 0) {
    float mu = s1[0] / (float)Mtot;
    float var = s2[0] / (float)Mtot - mu * mu;
    stats[c] = mu;
    stats[O + c] = rsqrtf(var + 1e-5f);
  }
}

__global__ __launch_bounds__(256) void bn_norm(
    const float* __restrict__ y, const float* __restrict__ stats,
    const float* __restrict__ gam, const float* __restrict__ bet,
    _Float16* __restrict__ out, size_t total, int O) {
  size_t i = (size_t)blockIdx.x * 256 + threadIdx.x;
  if (i >= total) return;
  int c = (int)(i % O);
  float v = (y[i] - stats[c]) * stats[O + c] * gam[c] + bet[c];
  out[i] = (_Float16)fmaxf(v, 0.f);
}

// ---------------------------------------------------------------------------
// Register-blocked WMMA GEMM: out[M][N] = act(A[M][K] @ W[N][K]^T + bias).
// Each wave: 1 M-tile x 4 N-tiles -> one A fragment feeds 4 WMMAs per k-step.
// Requires N % 64 == 0 (all call sites: 512, 256, 128, 64).
// ---------------------------------------------------------------------------
__global__ __launch_bounds__(256) void gemm_wmma4(
    const _Float16* __restrict__ A, const _Float16* __restrict__ W,
    const float* __restrict__ bias, _Float16* __restrict__ out,
    int M, int N, int K, int lda, int ldout, int relu) {
  int wave = (blockIdx.x * 256 + threadIdx.x) >> 5;
  int lane = threadIdx.x & 31;
  int ng = N >> 6;  // groups of 4 N-tiles
  int mtile = wave / ng, grp = wave % ng;
  if (mtile * 16 >= M) return;
  int n0 = grp * 64;
  v8f acc0 = {}, acc1 = {}, acc2 = {}, acc3 = {};
  for (int k0 = 0; k0 < K; k0 += 32) {
    if (k0 + 64 < K) {
      __builtin_prefetch(
          (const void*)(A + (size_t)(mtile * 16 + (lane & 15)) * lda + k0 + 64),
          0, 1);
    }
    v16h a = load_a16(A, lda, mtile * 16, k0, lane);
    v16h b0 = load_b16(W, K, n0 +  0, k0, lane);
    v16h b1 = load_b16(W, K, n0 + 16, k0, lane);
    v16h b2 = load_b16(W, K, n0 + 32, k0, lane);
    v16h b3 = load_b16(W, K, n0 + 48, k0, lane);
    acc0 = wmma_16x16x32(a, b0, acc0);
    acc1 = wmma_16x16x32(a, b1, acc1);
    acc2 = wmma_16x16x32(a, b2, acc2);
    acc3 = wmma_16x16x32(a, b3, acc3);
  }
  int g = lane >> 4;
  int nl = lane & 15;
  v8f accs[4] = {acc0, acc1, acc2, acc3};
#pragma unroll
  for (int q = 0; q < 4; ++q) {
    int n = n0 + q * 16 + nl;
    float bs = bias ? bias[n] : 0.f;
#pragma unroll
    for (int r = 0; r < 8; ++r) {
      int m = mtile * 16 + g * 8 + r;
      float v = accs[q][r] + bs;
      if (relu) v = fmaxf(v, 0.f);
      out[(size_t)m * ldout + n] = (_Float16)v;
    }
  }
}

// ---------------------------------------------------------------------------
// Coordinate embedder: emb = relu(input_seq @ mw^T + mb), [512][32] f16
// ---------------------------------------------------------------------------
__global__ __launch_bounds__(256) void emb_kernel(
    const float* __restrict__ input_seq, const float* __restrict__ mw,
    const float* __restrict__ mb, _Float16* __restrict__ emb) {
  int idx = blockIdx.x * 256 + threadIdx.x;
  if (idx >= NCROP * 32) return;
  int bs = idx >> 5, j = idx & 31;
  float y = input_seq[(size_t)bs * 2 + 0];
  float x = input_seq[(size_t)bs * 2 + 1];
  float v = mw[j * 2 + 0] * y + mw[j * 2 + 1] * x + mb[j];
  emb[idx] = (_Float16)fmaxf(v, 0.f);
}

// ---------------------------------------------------------------------------
// Condition encoder: relu([cond|steps] @ ew1^T + eb1) @ ew2^T + eb2
// writes f16 into combined[:, 512:576]
// ---------------------------------------------------------------------------
__global__ __launch_bounds__(128) void cond_kernel(
    const float* __restrict__ conditions, const float* __restrict__ steps,
    const float* __restrict__ ew1, const float* __restrict__ eb1,
    const float* __restrict__ ew2, const float* __restrict__ eb2,
    _Float16* __restrict__ combined, int ldcomb) {
  __shared__ float enh[6];
  __shared__ float hid[128];
  int bs = blockIdx.x;
  int b = bs >> 6;
  if (threadIdx.x < 5) enh[threadIdx.x] = conditions[b * 5 + threadIdx.x];
  if (threadIdx.x == 5) enh[5] = steps[bs];
  __syncthreads();
  int j = threadIdx.x;
  float a = eb1[j];
#pragma unroll
  for (int k = 0; k < 6; ++k) a += ew1[j * 6 + k] * enh[k];
  hid[j] = fmaxf(a, 0.f);
  __syncthreads();
  if (j < 64) {
    float o = eb2[j];
    for (int k = 0; k < 128; ++k) o += ew2[j * 128 + k] * hid[k];
    combined[(size_t)bs * ldcomb + 512 + j] = (_Float16)o;
  }
}

// ---------------------------------------------------------------------------
// 2-layer LSTM (hidden 256, batch 8 padded to 16) in one workgroup (16 waves).
// Gate GEMMs: [16 x K] x [K x 1024] via WMMA with 4-wide N register blocking
// (one LDS A fragment feeds 4 WMMAs). h/c state in LDS; layer-1 hidden states
// written as f16 into combined[:, 0:256].
// ---------------------------------------------------------------------------
__global__ __launch_bounds__(512) void lstm_kernel(
    const _Float16* __restrict__ emb,
    const _Float16* __restrict__ wih0, const _Float16* __restrict__ whh0,
    const float* __restrict__ bih0, const float* __restrict__ bhh0,
    const _Float16* __restrict__ wih1, const _Float16* __restrict__ whh1,
    const float* __restrict__ bih1, const float* __restrict__ bhh1,
    _Float16* __restrict__ combined, int ldcomb) {
  __shared__ alignas(16) _Float16 xsh[16 * 32];     // layer-0 input tile
  __shared__ alignas(16) _Float16 h0h[16 * 256];    // layer-0 hidden (f16)
  __shared__ alignas(16) _Float16 h1h[16 * 256];    // layer-1 hidden (f16)
  __shared__ float c0s[8 * 256];
  __shared__ float c1s[8 * 256];
  __shared__ alignas(16) _Float16 zsh[8 * 1024];    // gate pre-activations

  int tid = threadIdx.x;
  int lane = tid & 31;
  int wv = tid >> 5;       // 16 waves
  int nbase = wv * 64;     // 4 N-tiles per wave covering N = 1024

  for (int i = tid; i < 16 * 256; i += 512) { h0h[i] = (_Float16)0.f; h1h[i] = (_Float16)0.f; }
  for (int i = tid; i < 8 * 256; i += 512)  { c0s[i] = 0.f; c1s[i] = 0.f; }
  for (int i = tid; i < 16 * 32; i += 512)  xsh[i] = (_Float16)0.f;
  __syncthreads();

  for (int t = 0; t < SEQ; ++t) {
    // stage x_t (rows 0..7; rows 8..15 stay zero)
    for (int i = tid; i < 8 * 32; i += 512) {
      int b = i >> 5, j = i & 31;
      xsh[b * 32 + j] = emb[((size_t)b * SEQ + t) * 32 + j];
    }
    __syncthreads();

    // ---- layer 0 gate GEMM: z = x@Wih0^T + h0@Whh0^T + bias ----
    {
      v8f acc[4] = {};
      v16h ax = load_a16(xsh, 32, 0, 0, lane);
#pragma unroll
      for (int q = 0; q < 4; ++q) {
        v16h b = load_b16(wih0, 32, nbase + q * 16, 0, lane);
        acc[q] = wmma_16x16x32(ax, b, acc[q]);
      }
      for (int k0 = 0; k0 < 256; k0 += 32) {
        v16h a = load_a16(h0h, 256, 0, k0, lane);
#pragma unroll
        for (int q = 0; q < 4; ++q) {
          v16h b = load_b16(whh0, 256, nbase + q * 16, k0, lane);
          acc[q] = wmma_16x16x32(a, b, acc[q]);
        }
      }
      int g = lane >> 4;
#pragma unroll
      for (int q = 0; q < 4; ++q) {
        int n = nbase + q * 16 + (lane & 15);
        float bz = bih0[n] + bhh0[n];
#pragma unroll
        for (int r = 0; r < 8; ++r) {
          int m = g * 8 + r;
          if (m < 8) zsh[m * 1024 + n] = (_Float16)(acc[q][r] + bz);
        }
      }
    }
    __syncthreads();

    // ---- layer 0 gates ----
    for (int i = tid; i < 8 * 256; i += 512) {
      int b = i >> 8, j = i & 255;
      float zi = (float)zsh[b * 1024 + j];
      float zf = (float)zsh[b * 1024 + 256 + j];
      float zg = (float)zsh[b * 1024 + 512 + j];
      float zo = (float)zsh[b * 1024 + 768 + j];
      float cc = sigmoidf(zf) * c0s[i] + sigmoidf(zi) * tanhf(zg);
      float hh = sigmoidf(zo) * tanhf(cc);
      c0s[i] = cc;
      h0h[b * 256 + j] = (_Float16)hh;
    }
    __syncthreads();

    // ---- layer 1 gate GEMM: z = h0@Wih1^T + h1@Whh1^T + bias ----
    {
      v8f acc[4] = {};
      for (int k0 = 0; k0 < 256; k0 += 32) {
        v16h a = load_a16(h0h, 256, 0, k0, lane);
#pragma unroll
        for (int q = 0; q < 4; ++q) {
          v16h b = load_b16(wih1, 256, nbase + q * 16, k0, lane);
          acc[q] = wmma_16x16x32(a, b, acc[q]);
        }
      }
      for (int k0 = 0; k0 < 256; k0 += 32) {
        v16h a = load_a16(h1h, 256, 0, k0, lane);
#pragma unroll
        for (int q = 0; q < 4; ++q) {
          v16h b = load_b16(whh1, 256, nbase + q * 16, k0, lane);
          acc[q] = wmma_16x16x32(a, b, acc[q]);
        }
      }
      int g = lane >> 4;
#pragma unroll
      for (int q = 0; q < 4; ++q) {
        int n = nbase + q * 16 + (lane & 15);
        float bz = bih1[n] + bhh1[n];
#pragma unroll
        for (int r = 0; r < 8; ++r) {
          int m = g * 8 + r;
          if (m < 8) zsh[m * 1024 + n] = (_Float16)(acc[q][r] + bz);
        }
      }
    }
    __syncthreads();

    // ---- layer 1 gates + write lstm_out into combined[:, 0:256] ----
    for (int i = tid; i < 8 * 256; i += 512) {
      int b = i >> 8, j = i & 255;
      float zi = (float)zsh[b * 1024 + j];
      float zf = (float)zsh[b * 1024 + 256 + j];
      float zg = (float)zsh[b * 1024 + 512 + j];
      float zo = (float)zsh[b * 1024 + 768 + j];
      float cc = sigmoidf(zf) * c1s[i] + sigmoidf(zi) * tanhf(zg);
      float hh = sigmoidf(zo) * tanhf(cc);
      c1s[i] = cc;
      h1h[b * 256 + j] = (_Float16)hh;
      combined[((size_t)b * SEQ + t) * ldcomb + j] = (_Float16)hh;
    }
    __syncthreads();
  }
}

// ---------------------------------------------------------------------------
// Final linear head: out[512][2] = ctx2 @ ow^T + ob   (f32 output)
// ---------------------------------------------------------------------------
__global__ __launch_bounds__(256) void out_kernel(
    const _Float16* __restrict__ ctx2, const float* __restrict__ ow,
    const float* __restrict__ ob, float* __restrict__ out) {
  int idx = blockIdx.x * 256 + threadIdx.x;
  if (idx >= NCROP * 2) return;
  int bs = idx >> 1, k = idx & 1;
  float a = ob[k];
#pragma unroll 4
  for (int j = 0; j < 64; ++j)
    a += (float)ctx2[(size_t)bs * 64 + j] * ow[k * 64 + j];
  out[idx] = a;
}

// ---------------------------------------------------------------------------
// Host orchestration
// ---------------------------------------------------------------------------
extern "C" void kernel_launch(void* const* d_in, const int* in_sizes, int n_in,
                              void* d_out, int out_size, void* d_ws,
                              size_t ws_size, hipStream_t stream) {
  (void)in_sizes; (void)n_in; (void)out_size; (void)ws_size;

  const float* full_map   = (const float*)d_in[0];
  const float* conditions = (const float*)d_in[1];
  const float* input_seq  = (const float*)d_in[2];
  const float* steps_rem  = (const float*)d_in[3];
  const float* cw1 = (const float*)d_in[4],  *cb1 = (const float*)d_in[5];
  const float* g1  = (const float*)d_in[6],  *be1 = (const float*)d_in[7];
  const float* cw2 = (const float*)d_in[8],  *cb2 = (const float*)d_in[9];
  const float* g2  = (const float*)d_in[10], *be2 = (const float*)d_in[11];
  const float* cw3 = (const float*)d_in[12], *cb3 = (const float*)d_in[13];
  const float* g3  = (const float*)d_in[14], *be3 = (const float*)d_in[15];
  const float* cw4 = (const float*)d_in[16], *cb4 = (const float*)d_in[17];
  const float* g4  = (const float*)d_in[18], *be4 = (const float*)d_in[19];
  const float* fw1 = (const float*)d_in[20], *fb1 = (const float*)d_in[21];
  const float* fw2 = (const float*)d_in[22], *fb2 = (const float*)d_in[23];
  const float* ew1 = (const float*)d_in[24], *eb1 = (const float*)d_in[25];
  const float* ew2 = (const float*)d_in[26], *eb2 = (const float*)d_in[27];
  const float* mw  = (const float*)d_in[28], *mb  = (const float*)d_in[29];
  const float* wih0 = (const float*)d_in[30], *whh0 = (const float*)d_in[31];
  const float* bih0 = (const float*)d_in[32], *bhh0 = (const float*)d_in[33];
  const float* wih1 = (const float*)d_in[34], *whh1 = (const float*)d_in[35];
  const float* bih1 = (const float*)d_in[36], *bhh1 = (const float*)d_in[37];
  const float* xw1 = (const float*)d_in[38], *xb1 = (const float*)d_in[39];
  const float* xw2 = (const float*)d_in[40], *xb2 = (const float*)d_in[41];
  const float* ow  = (const float*)d_in[42], *ob  = (const float*)d_in[43];
  float* out = (float*)d_out;

  // ---- workspace carve-up (256B aligned) ----
  char* wsb = (char*)d_ws;
  size_t off = 0;
  auto alloc = [&](size_t bytes) -> void* {
    void* p = wsb + off;
    off = (off + bytes + 255) & ~(size_t)255;
    return p;
  };
  _Float16* crops   = (_Float16*)alloc((size_t)NCROP * 128 * 128 * 3 * 2);
  float*    raw     = (float*)   alloc((size_t)NCROP * 64 * 64 * 16 * 4);  // reused all convs
  _Float16* nrm     = (_Float16*)alloc((size_t)NCROP * 64 * 64 * 16 * 2);  // reused all convs
  float*    stats   = (float*)   alloc(2 * 128 * 4);
  float*    partial = (float*)   alloc((size_t)256 * 128 * 2 * 4);
  _Float16* wpc1    = (_Float16*)alloc((size_t)16 * 32 * 2);
  _Float16* wpc2    = (_Float16*)alloc((size_t)32 * 160 * 2);
  _Float16* wpc3    = (_Float16*)alloc((size_t)64 * 288 * 2);
  _Float16* wpc4    = (_Float16*)alloc((size_t)128 * 576 * 2);
  _Float16* wpfc1   = (_Float16*)alloc((size_t)512 * 8192 * 2);
  _Float16* wpfc2   = (_Float16*)alloc((size_t)256 * 512 * 2);
  _Float16* wpih0   = (_Float16*)alloc((size_t)1024 * 32 * 2);
  _Float16* wphh0   = (_Float16*)alloc((size_t)1024 * 256 * 2);
  _Float16* wpih1   = (_Float16*)alloc((size_t)1024 * 256 * 2);
  _Float16* wphh1   = (_Float16*)alloc((size_t)1024 * 256 * 2);
  _Float16* wpx1    = (_Float16*)alloc((size_t)128 * 576 * 2);
  _Float16* wpx2    = (_Float16*)alloc((size_t)64 * 128 * 2);
  _Float16* embb    = (_Float16*)alloc((size_t)NCROP * 32 * 2);
  _Float16* combined= (_Float16*)alloc((size_t)NCROP * 576 * 2);
  _Float16* fc1out  = (_Float16*)alloc((size_t)NCROP * 512 * 2);
  _Float16* ctx1    = (_Float16*)alloc((size_t)NCROP * 128 * 2);
  _Float16* ctx2    = (_Float16*)alloc((size_t)NCROP * 64 * 2);

  // ---- weight prep (deterministic, capture-safe) ----
  prep_conv_w<3, 16>  <<<(16 * 32 + 255) / 256, 256, 0, stream>>>(cw1, wpc1);
  prep_conv_w<16, 32> <<<(32 * 160 + 255) / 256, 256, 0, stream>>>(cw2, wpc2);
  prep_conv_w<32, 64> <<<(64 * 288 + 255) / 256, 256, 0, stream>>>(cw3, wpc3);
  prep_conv_w<64, 128><<<(128 * 576 + 255) / 256, 256, 0, stream>>>(cw4, wpc4);
  prep_fc1<<<(512 * 8192 + 255) / 256, 256, 0, stream>>>(fw1, wpfc1);
  cast_f16_kernel<<<(256 * 512 + 255) / 256, 256, 0, stream>>>(fw2, wpfc2, 256 * 512);
  cast_f16_kernel<<<(1024 * 32 + 255) / 256, 256, 0, stream>>>(wih0, wpih0, 1024 * 32);
  cast_f16_kernel<<<(1024 * 256 + 255) / 256, 256, 0, stream>>>(whh0, wphh0, 1024 * 256);
  cast_f16_kernel<<<(1024 * 256 + 255) / 256, 256, 0, stream>>>(wih1, wpih1, 1024 * 256);
  cast_f16_kernel<<<(1024 * 256 + 255) / 256, 256, 0, stream>>>(whh1, wphh1, 1024 * 256);
  cast_f16_kernel<<<(128 * 576 + 255) / 256, 256, 0, stream>>>(xw1, wpx1, 128 * 576);
  cast_f16_kernel<<<(64 * 128 + 255) / 256, 256, 0, stream>>>(xw2, wpx2, 64 * 128);

  // ---- crop gather ----
  crop_kernel<<<(int)(((size_t)NCROP * 128 * 128 * 3 + 255) / 256), 256, 0,
                stream>>>(full_map, input_seq, crops);

  // ---- conv1: 128x128x3 -> 64x64x16  (Mtiles = 131072 waves) ----
  conv_wmma<128, 3, 16><<<16384, 256, 0, stream>>>(crops, wpc1, cb1, raw);
  bn_stats_partial<<<dim3(256, 16), 256, 0, stream>>>(raw, 512 * 64 * 64, 16, partial);
  bn_stats_final<<<16, 256, 0, stream>>>(partial, 16, 512 * 64 * 64, stats);
  bn_norm<<<(int)(((size_t)512 * 64 * 64 * 16 + 255) / 256), 256, 0, stream>>>(
      raw, stats, g1, be1, nrm, (size_t)512 * 64 * 64 * 16, 16);

  // ---- conv2: 64x64x16 -> 32x32x32  (32768 waves) ----
  conv_wmma<64, 16, 32><<<4096, 256, 0, stream>>>(nrm, wpc2, cb2, raw);
  bn_stats_partial<<<dim3(256, 32), 256, 0, stream>>>(raw, 512 * 32 * 32, 32, partial);
  bn_stats_final<<<32, 256, 0, stream>>>(partial, 32, 512 * 32 * 32, stats);
  bn_norm<<<(int)(((size_t)512 * 32 * 32 * 32 + 255) / 256), 256, 0, stream>>>(
      raw, stats, g2, be2, nrm, (size_t)512 * 32 * 32 * 32, 32);

  // ---- conv3: 32x32x32 -> 16x16x64  (8192 waves) ----
  conv_wmma<32, 32, 64><<<1024, 256, 0, stream>>>(nrm, wpc3, cb3, raw);
  bn_stats_partial<<<dim3(256, 64), 256, 0, stream>>>(raw, 512 * 16 * 16, 64, partial);
  bn_stats_final<<<64, 256, 0, stream>>>(partial, 64, 512 * 16 * 16, stats);
  bn_norm<<<(int)(((size_t)512 * 16 * 16 * 64 + 255) / 256), 256, 0, stream>>>(
      raw, stats, g3, be3, nrm, (size_t)512 * 16 * 16 * 64, 64);

  // ---- conv4: 16x16x64 -> 8x8x128  (2048 waves) ----
  conv_wmma<16, 64, 128><<<256, 256, 0, stream>>>(nrm, wpc4, cb4, raw);
  bn_stats_partial<<<dim3(256, 128), 256, 0, stream>>>(raw, 512 * 8 * 8, 128, partial);
  bn_stats_final<<<128, 256, 0, stream>>>(partial, 128, 512 * 8 * 8, stats);
  bn_norm<<<(int)(((size_t)512 * 8 * 8 * 128 + 255) / 256), 256, 0, stream>>>(
      raw, stats, g4, be4, nrm, (size_t)512 * 8 * 8 * 128, 128);

  // ---- FC head: 8192 -> 512 (relu) -> 256, latter into combined[:,256:512] ----
  gemm_wmma4<<<32, 256, 0, stream>>>(nrm, wpfc1, fb1, fc1out,
                                     NCROP, 512, 8192, 8192, 512, 1);
  gemm_wmma4<<<16, 256, 0, stream>>>(fc1out, wpfc2, fb2, combined + 256,
                                     NCROP, 256, 512, 512, 576, 0);

  // ---- sequence path ----
  emb_kernel<<<(NCROP * 32 + 255) / 256, 256, 0, stream>>>(input_seq, mw, mb, embb);
  lstm_kernel<<<1, 512, 0, stream>>>(embb, wpih0, wphh0, bih0, bhh0,
                                     wpih1, wphh1, bih1, bhh1, combined, 576);
  cond_kernel<<<NCROP, 128, 0, stream>>>(conditions, steps_rem, ew1, eb1,
                                         ew2, eb2, combined, 576);

  // ---- context MLP (576 -> 128 -> 64, both relu) + output head ----
  gemm_wmma4<<<8, 256, 0, stream>>>(combined, wpx1, xb1, ctx1,
                                    NCROP, 128, 576, 576, 128, 1);
  gemm_wmma4<<<4, 256, 0, stream>>>(ctx1, wpx2, xb2, ctx2,
                                    NCROP, 64, 128, 128, 64, 1);
  out_kernel<<<(NCROP * 2 + 255) / 256, 256, 0, stream>>>(ctx2, ow, ob, out);
}
